// ComplexFFTRadNet_72808285601854
// MI455X (gfx1250) — compile-verified, matrix operand order checked by reading, not run on hardware
//
#include <hip/hip_runtime.h>
#include <hip/hip_bf16.h>
#include <cstdint>

typedef _Float16 v16h __attribute__((ext_vector_type(16)));
typedef _Float16 v8h  __attribute__((ext_vector_type(8)));
typedef float    v8f  __attribute__((ext_vector_type(8)));

// Problem geometry (fixed by the reference)
constexpr int NB = 4, HH = 128, WW = 224;
constexpr int PP = NB * HH * WW;          // 114688 output positions
constexpr float BN_EPS = 1e-5f;

__device__ __forceinline__ v8f wmma_f16(v16h a, v16h b, v8f c) {
  // D = A(16x32 f16) * B(32x16 f16) + C(16x16 f32)
  return __builtin_amdgcn_wmma_f32_16x16x32_f16(false, a, false, b, (short)0, c,
                                                false, false);
}

// CDNA5 async global->LDS copy: 32 bytes per call-site pair, ASYNCcnt-tracked.
// LDS dest offset and global source share INST_OFFSET, so one address pair
// covers both b128 transfers.
__device__ __forceinline__ void async_copy_32B(uint32_t lds_off, const void* gptr) {
  const uint64_t ga = (uint64_t)(uintptr_t)gptr;
  asm volatile("global_load_async_to_lds_b128 %0, %1, off"
               :: "v"(lds_off), "v"(ga) : "memory");
  asm volatile("global_load_async_to_lds_b128 %0, %1, off offset:16"
               :: "v"(lds_off), "v"(ga) : "memory");
}
__device__ __forceinline__ void wait_async0() {
  asm volatile("s_wait_asynccnt 0x0" ::: "memory");
}

// ---------------------------------------------------------------------------
// Implicit-GEMM complex conv, channels-last fp16.
//   xr/xi : [PP][Cin] fp16 activations (xi ignored when Ktap == Cin)
//   wpr   : packed [9][O][Ktap] fp16 = [wr ; -wi]  (real-output weights)
//   wpi   : packed [9][O][Ktap] fp16 = [wi ;  wr]  (imag-output weights)
//   yr/yi : [PP][O] fp16 conv outputs
// Tile: 128(M,pos) x 64(N,oc) x 32(K) per block, 8 waves, double-buffered LDS
// filled by CDNA5 async global->LDS loads.
// ---------------------------------------------------------------------------
__global__ __launch_bounds__(256)
void conv_cplx_wmma(const _Float16* __restrict__ xr,
                    const _Float16* __restrict__ xi,
                    const _Float16* __restrict__ wpr,
                    const _Float16* __restrict__ wpi,
                    _Float16* __restrict__ yr,
                    _Float16* __restrict__ yi,
                    int Cin, int Ktap, int O)
{
  constexpr int BM = 128, BN = 64, BK = 32, LDA = 40, LDB = 40;
  __shared__ _Float16 sA[2][BM * LDA];       // [pos][k]
  __shared__ _Float16 sB[2][2 * BN * LDB];   // [{Br,Bi} x oc][k]

  const int t       = threadIdx.x;
  const int lane    = t & 31;
  const int lane_lo = lane & 15;
  const int hi      = lane >> 4;
  const int wid     = t >> 5;
  const int wave_m  = wid & 3;    // 4 M-groups of 32 rows
  const int wave_n  = wid >> 2;   // 2 N-groups of 32 cols
  const int pos_base = blockIdx.x * BM;
  const int oc_base  = blockIdx.y * BN;

  // ---- cooperative loader coordinates (2 threads per 32-wide row) ----
  const int arow = t >> 1;            // 0..127: A row / B row index
  const int kseg = (t & 1) * 16;      // 16-half segment within the K-chunk
  const int p    = pos_base + arow;
  const int pn   = p / (HH * WW);
  const int prm  = p % (HH * WW);
  const int ph   = prm / WW, pw = prm % WW;
  const int bmat = arow >> 6;         // 0 = Br, 1 = Bi
  const int bo   = arow & 63;
  const int oc_ld = oc_base + bo;

  const int nk = Ktap / BK;           // K-steps per tap (Ktap always %32==0)
  const int KT = 9 * nk;              // total K-steps over all 9 taps

  const v8h vz = {};

  // Issue the (A,B) tile for K-step `kt` into LDS buffer `buf`.
  // Valid lanes use async global->LDS b128 loads; padding lanes zero-fill
  // through normal ds_stores (DScnt, ordered before the barrier by codegen).
  auto issue_tile = [&](int kt, int buf) {
    const int tap = kt / nk;
    const int ks  = (kt - tap * nk) * BK;
    const int kk  = ks + kseg;                 // K index (16-aligned)
    const int dh  = tap / 3 - 1, dw = tap % 3 - 1;
    const int ih  = ph + dh, iw = pw + dw;
    // A: im2col patch (xr half then xi half; Cin %16==0 so a 16-half segment
    // never straddles the real/imag boundary)
    _Float16* aSlot = &sA[buf][arow * LDA + kseg];
    if ((unsigned)ih < (unsigned)HH && (unsigned)iw < (unsigned)WW) {
      const bool half = (kk >= Cin);
      const int  ci   = kk - (half ? Cin : 0);
      const _Float16* src = half ? xi : xr;
      const _Float16* g =
          src + ((size_t)((pn * HH + ih) * WW + iw)) * (size_t)Cin + ci;
      async_copy_32B((uint32_t)(uintptr_t)aSlot, g);
    } else {
      *(v8h*)aSlot       = vz;
      *(v8h*)(aSlot + 8) = vz;
    }
    // B: packed weights, [tap][oc][k] so the K run is contiguous
    _Float16* bSlot = &sB[buf][(bmat * BN + bo) * LDB + kseg];
    if (oc_ld < O) {
      const _Float16* wsrc = bmat ? wpi : wpr;
      const _Float16* g = wsrc + ((size_t)tap * O + oc_ld) * (size_t)Ktap + kk;
      async_copy_32B((uint32_t)(uintptr_t)bSlot, g);
    } else {
      *(v8h*)bSlot       = vz;
      *(v8h*)(bSlot + 8) = vz;
    }
  };

  // ---- WMMA fragment loads per documented 16-bit layouts ----
  // lanes 0..15 hold K {0..7,16..23}; lanes 16..31 hold K {8..15,24..31}
  const int kb = hi * 8;
  auto fragA = [&](int buf, int mrow) -> v16h {
    const _Float16* q = &sA[buf][(mrow + lane_lo) * LDA + kb];
    v8h lo = *(const v8h*)q;
    v8h hp = *(const v8h*)(q + 16);
    return __builtin_shufflevector(lo, hp, 0, 1, 2, 3, 4, 5, 6, 7,
                                           8, 9, 10, 11, 12, 13, 14, 15);
  };
  auto fragB = [&](int buf, int mat, int nrow) -> v16h {
    const _Float16* q = &sB[buf][(mat * BN + nrow + lane_lo) * LDB + kb];
    v8h lo = *(const v8h*)q;
    v8h hp = *(const v8h*)(q + 16);
    return __builtin_shufflevector(lo, hp, 0, 1, 2, 3, 4, 5, 6, 7,
                                           8, 9, 10, 11, 12, 13, 14, 15);
  };

  v8f aR[2][2] = {};   // real-output accumulators (2 M-tiles x 2 N-tiles)
  v8f aI[2][2] = {};   // imag-output accumulators

  issue_tile(0, 0);
  wait_async0();
  __syncthreads();

  for (int kt = 0; kt < KT; ++kt) {
    const int cur = kt & 1;
    if (kt + 1 < KT) issue_tile(kt + 1, cur ^ 1);   // async prefetch next tile

    v16h a0  = fragA(cur, wave_m * 32);
    v16h a1  = fragA(cur, wave_m * 32 + 16);
    v16h br0 = fragB(cur, 0, wave_n * 32);
    v16h br1 = fragB(cur, 0, wave_n * 32 + 16);
    v16h bi0 = fragB(cur, 1, wave_n * 32);
    v16h bi1 = fragB(cur, 1, wave_n * 32 + 16);

    aR[0][0] = wmma_f16(a0, br0, aR[0][0]);
    aR[0][1] = wmma_f16(a0, br1, aR[0][1]);
    aR[1][0] = wmma_f16(a1, br0, aR[1][0]);
    aR[1][1] = wmma_f16(a1, br1, aR[1][1]);
    aI[0][0] = wmma_f16(a0, bi0, aI[0][0]);
    aI[0][1] = wmma_f16(a0, bi1, aI[0][1]);
    aI[1][0] = wmma_f16(a1, bi0, aI[1][0]);
    aI[1][1] = wmma_f16(a1, bi1, aI[1][1]);

    if (kt + 1 < KT) wait_async0();   // next tile resident before the barrier
    __syncthreads();
  }

  // ---- store: D VGPR r -> (M = r + hi*8, N = lane_lo) ----
  for (int mi = 0; mi < 2; ++mi)
    for (int ni = 0; ni < 2; ++ni) {
      const int nc = oc_base + wave_n * 32 + ni * 16 + lane_lo;
      if (nc >= O) continue;
      const int mb = pos_base + wave_m * 32 + mi * 16 + hi * 8;
#pragma unroll
      for (int r = 0; r < 8; ++r) {
        const size_t idx = (size_t)(mb + r) * (size_t)O + nc;
        yr[idx] = (_Float16)aR[mi][ni][r];
        yi[idx] = (_Float16)aI[mi][ni][r];
      }
    }
}

// ---------------------------------------------------------------------------
// NCHW fp32 -> channels-last fp16
// ---------------------------------------------------------------------------
__global__ void pack_x_kernel(const float* __restrict__ x,
                              _Float16* __restrict__ o, int C)
{
  const size_t total = (size_t)PP * C;
  for (size_t idx = (size_t)blockIdx.x * blockDim.x + threadIdx.x; idx < total;
       idx += (size_t)gridDim.x * blockDim.x) {
    const int c = (int)(idx % C);
    const int p = (int)(idx / C);
    const int n = p / (HH * WW);
    const int rem = p % (HH * WW);
    const int h = rem / WW, w = rem % WW;
    o[idx] = (_Float16)x[(((size_t)n * C + c) * HH + h) * WW + w];
  }
}

// ---------------------------------------------------------------------------
// Pack [O][I][3][3] fp32 complex weight pair into two fp16 GEMM B matrices:
//   pr[tap][o][k] = k<I ? wr : -wi     pi[tap][o][k] = k<I ? wi : wr
// dual==0 (layer 1, real input): Ktap = I, pr = wr, pi = wi.
// ---------------------------------------------------------------------------
__global__ void pack_w_kernel(const float* __restrict__ wr,
                              const float* __restrict__ wi,
                              _Float16* __restrict__ pr,
                              _Float16* __restrict__ pi,
                              int O, int I, int dual)
{
  const int Ktap = dual ? 2 * I : I;
  const size_t total = (size_t)9 * O * Ktap;
  for (size_t idx = (size_t)blockIdx.x * blockDim.x + threadIdx.x; idx < total;
       idx += (size_t)gridDim.x * blockDim.x) {
    const int k = (int)(idx % Ktap);
    const size_t r1 = idx / Ktap;
    const int o = (int)(r1 % O);
    const int tap = (int)(r1 / O);
    float vr, vi;
    if (k < I) {
      vr = wr[((size_t)o * I + k) * 9 + tap];
      vi = wi[((size_t)o * I + k) * 9 + tap];
    } else {
      const int k2 = k - I;
      vr = -wi[((size_t)o * I + k2) * 9 + tap];
      vi =  wr[((size_t)o * I + k2) * 9 + tap];
    }
    pr[idx] = (_Float16)vr;
    pi[idx] = (_Float16)vi;
  }
}

// ---------------------------------------------------------------------------
// Per-channel (sum, sumsq) for real and imag parts -> stats[(comp*C+c)*2 + {0,1}]
// LDS ds_add_f32 reduction, then global_atomic_add_f32 flush.
// ---------------------------------------------------------------------------
__global__ void stats_kernel(const _Float16* __restrict__ br,
                             const _Float16* __restrict__ bi,
                             float* __restrict__ stats, int C)
{
  __shared__ float s[2 * 144 * 2];
  const int nslots = 2 * C * 2;
  for (int j = threadIdx.x; j < nslots; j += blockDim.x) s[j] = 0.f;
  __syncthreads();
  const size_t PC = (size_t)PP * C;
  const size_t total = 2 * PC;
  for (size_t idx = (size_t)blockIdx.x * blockDim.x + threadIdx.x; idx < total;
       idx += (size_t)gridDim.x * blockDim.x) {
    const int comp = idx >= PC;
    const size_t rem = idx - (comp ? PC : 0);
    const int c = (int)(rem % C);
    const float v = (float)(comp ? bi[rem] : br[rem]);
    atomicAdd(&s[(comp * C + c) * 2],     v);
    atomicAdd(&s[(comp * C + c) * 2 + 1], v * v);
  }
  __syncthreads();
  for (int j = threadIdx.x; j < nslots; j += blockDim.x)
    atomicAdd(&stats[j], s[j]);
}

// ---------------------------------------------------------------------------
// In-place BN (training-style stats over N,H,W) + shared affine + ReLU
// ---------------------------------------------------------------------------
__global__ void bn_apply_kernel(_Float16* __restrict__ br,
                                _Float16* __restrict__ bi,
                                const float* __restrict__ stats,
                                const float* __restrict__ g,
                                const float* __restrict__ b, int C)
{
  const size_t PC = (size_t)PP * C;
  const size_t total = 2 * PC;
  const float invP = 1.f / (float)PP;
  for (size_t idx = (size_t)blockIdx.x * blockDim.x + threadIdx.x; idx < total;
       idx += (size_t)gridDim.x * blockDim.x) {
    const int comp = idx >= PC;
    const size_t rem = idx - (comp ? PC : 0);
    const int c = (int)(rem % C);
    const float sum = stats[(comp * C + c) * 2];
    const float sq  = stats[(comp * C + c) * 2 + 1];
    const float mu  = sum * invP;
    const float var = sq * invP - mu * mu;
    const float rs  = rsqrtf(var + BN_EPS);
    _Float16* buf = comp ? bi : br;
    float v = (float)buf[rem];
    v = (v - mu) * rs * g[c] + b[c];
    buf[rem] = (_Float16)(v > 0.f ? v : 0.f);
  }
}

// ---------------------------------------------------------------------------
// mag = sqrt(r^2 + i^2), channels-last fp16
// ---------------------------------------------------------------------------
__global__ void mag_kernel(const _Float16* __restrict__ r,
                           const _Float16* __restrict__ i,
                           _Float16* __restrict__ m, int C)
{
  const size_t total = (size_t)PP * C;
  for (size_t idx = (size_t)blockIdx.x * blockDim.x + threadIdx.x; idx < total;
       idx += (size_t)gridDim.x * blockDim.x) {
    const float a = (float)r[idx];
    const float b = (float)i[idx];
    m[idx] = (_Float16)sqrtf(a * a + b * b);
  }
}

// ---------------------------------------------------------------------------
// Head: 3x3 convs 96 -> {1 (sigmoid cls), 2 (reg)}; output NCHW fp32 concat.
// ---------------------------------------------------------------------------
__global__ __launch_bounds__(256)
void head_kernel(const _Float16* __restrict__ mag,
                 const float* __restrict__ wc, const float* __restrict__ bc,
                 const float* __restrict__ wg, const float* __restrict__ bg,
                 float* __restrict__ out)
{
  constexpr int C = 96, WOFF = 96 * 9;
  __shared__ float lw[3 * 96 * 9];
  for (int j = threadIdx.x; j < WOFF; j += blockDim.x)     lw[j]        = wc[j];
  for (int j = threadIdx.x; j < 2 * WOFF; j += blockDim.x) lw[WOFF + j] = wg[j];
  __syncthreads();

  const int p = blockIdx.x * blockDim.x + threadIdx.x;   // PP % 256 == 0
  const int n = p / (HH * WW);
  const int rem = p % (HH * WW);
  const int h = rem / WW, w = rem % WW;

  float a0 = bc[0], a1 = bg[0], a2 = bg[1];
  for (int tap = 0; tap < 9; ++tap) {
    const int ih = h + tap / 3 - 1, iw = w + tap % 3 - 1;
    if ((unsigned)ih >= (unsigned)HH || (unsigned)iw >= (unsigned)WW) continue;
    const _Float16* mrow = mag + ((size_t)(n * HH + ih) * WW + iw) * C;
    for (int c = 0; c < C; ++c) {
      const float m = (float)mrow[c];
      a0 += m * lw[c * 9 + tap];
      a1 += m * lw[WOFF + c * 9 + tap];
      a2 += m * lw[WOFF + WOFF + c * 9 + tap];
    }
  }
  const size_t base = (size_t)n * 3 * HH * WW + (size_t)h * WW + w;
  out[base]                        = 1.f / (1.f + __expf(-a0));
  out[base + (size_t)HH * WW]      = a1;
  out[base + (size_t)2 * HH * WW]  = a2;
}

// ---------------------------------------------------------------------------
extern "C" void kernel_launch(void* const* d_in, const int* in_sizes, int n_in,
                              void* d_out, int out_size, void* d_ws, size_t ws_size,
                              hipStream_t stream)
{
  (void)in_sizes; (void)n_in; (void)out_size; (void)ws_size;

  const float* x   = (const float*)d_in[0];
  const float* w1r = (const float*)d_in[1];
  const float* w1i = (const float*)d_in[2];
  const float* g1  = (const float*)d_in[3];
  const float* b1  = (const float*)d_in[4];
  const float* w2r = (const float*)d_in[5];
  const float* w2i = (const float*)d_in[6];
  const float* g2  = (const float*)d_in[7];
  const float* b2  = (const float*)d_in[8];
  const float* w3r = (const float*)d_in[9];
  const float* w3i = (const float*)d_in[10];
  const float* g3  = (const float*)d_in[11];
  const float* b3  = (const float*)d_in[12];
  const float* w4r = (const float*)d_in[13];
  const float* w4i = (const float*)d_in[14];
  const float* g4  = (const float*)d_in[15];
  const float* b4  = (const float*)d_in[16];
  const float* wc  = (const float*)d_in[17];
  const float* bc  = (const float*)d_in[18];
  const float* wg  = (const float*)d_in[19];
  const float* bg  = (const float*)d_in[20];
  float* out = (float*)d_out;

  // ---- workspace bump allocator (~194 MB total) ----
  uint8_t* ws = (uint8_t*)d_ws;
  size_t off = 0;
  auto alloc = [&](size_t bytes) -> uint8_t* {
    uint8_t* ptr = ws + off;
    off += (bytes + 255) & ~(size_t)255;
    return ptr;
  };
  _Float16* x16   = (_Float16*)alloc((size_t)PP * 256 * 2);  // reused as mag later
  _Float16* actAr = (_Float16*)alloc((size_t)PP * 144 * 2);
  _Float16* actAi = (_Float16*)alloc((size_t)PP * 144 * 2);
  _Float16* actBr = (_Float16*)alloc((size_t)PP * 144 * 2);
  _Float16* actBi = (_Float16*)alloc((size_t)PP * 144 * 2);
  _Float16* wp1r  = (_Float16*)alloc((size_t)9 * 144 * 256 * 2);
  _Float16* wp1i  = (_Float16*)alloc((size_t)9 * 144 * 256 * 2);
  _Float16* wp2r  = (_Float16*)alloc((size_t)9 * 96 * 288 * 2);
  _Float16* wp2i  = (_Float16*)alloc((size_t)9 * 96 * 288 * 2);
  _Float16* wp3r  = (_Float16*)alloc((size_t)9 * 96 * 192 * 2);
  _Float16* wp3i  = (_Float16*)alloc((size_t)9 * 96 * 192 * 2);
  _Float16* wp4r  = (_Float16*)alloc((size_t)9 * 96 * 192 * 2);
  _Float16* wp4i  = (_Float16*)alloc((size_t)9 * 96 * 192 * 2);
  float*    stats = (float*)alloc(4 * 2 * 144 * 2 * sizeof(float));
  _Float16* mag   = x16;   // alias: x16 only needed until conv1 runs

  hipMemsetAsync(stats, 0, 4 * 2 * 144 * 2 * sizeof(float), stream);

  // pack input + weights
  pack_x_kernel<<<4096, 256, 0, stream>>>(x, x16, 256);
  pack_w_kernel<<<1296, 256, 0, stream>>>(w1r, w1i, wp1r, wp1i, 144, 256, 0);
  pack_w_kernel<<< 972, 256, 0, stream>>>(w2r, w2i, wp2r, wp2i, 96, 144, 1);
  pack_w_kernel<<< 648, 256, 0, stream>>>(w3r, w3i, wp3r, wp3i, 96, 96, 1);
  pack_w_kernel<<< 648, 256, 0, stream>>>(w4r, w4i, wp4r, wp4i, 96, 96, 1);

  const dim3 blk(256);
  constexpr int SS = 2 * 144 * 2;   // stats slice stride (floats)

  // layer 1: real input only (Ktap = Cin = 256), O = 144
  conv_cplx_wmma<<<dim3(PP / 128, 3), blk, 0, stream>>>(
      x16, x16, wp1r, wp1i, actAr, actAi, 256, 256, 144);
  stats_kernel<<<1024, 256, 0, stream>>>(actAr, actAi, stats + 0 * SS, 144);
  bn_apply_kernel<<<4096, 256, 0, stream>>>(actAr, actAi, stats + 0 * SS, g1, b1, 144);

  // layer 2: Cin = 144, Ktap = 288, O = 96
  conv_cplx_wmma<<<dim3(PP / 128, 2), blk, 0, stream>>>(
      actAr, actAi, wp2r, wp2i, actBr, actBi, 144, 288, 96);
  stats_kernel<<<1024, 256, 0, stream>>>(actBr, actBi, stats + 1 * SS, 96);
  bn_apply_kernel<<<4096, 256, 0, stream>>>(actBr, actBi, stats + 1 * SS, g2, b2, 96);

  // layer 3: Cin = 96, Ktap = 192, O = 96
  conv_cplx_wmma<<<dim3(PP / 128, 2), blk, 0, stream>>>(
      actBr, actBi, wp3r, wp3i, actAr, actAi, 96, 192, 96);
  stats_kernel<<<1024, 256, 0, stream>>>(actAr, actAi, stats + 2 * SS, 96);
  bn_apply_kernel<<<4096, 256, 0, stream>>>(actAr, actAi, stats + 2 * SS, g3, b3, 96);

  // layer 4
  conv_cplx_wmma<<<dim3(PP / 128, 2), blk, 0, stream>>>(
      actAr, actAi, wp4r, wp4i, actBr, actBi, 96, 192, 96);
  stats_kernel<<<1024, 256, 0, stream>>>(actBr, actBi, stats + 3 * SS, 96);
  bn_apply_kernel<<<4096, 256, 0, stream>>>(actBr, actBi, stats + 3 * SS, g4, b4, 96);

  // magnitude + head
  mag_kernel<<<4096, 256, 0, stream>>>(actBr, actBi, mag, 96);
  head_kernel<<<PP / 256, 256, 0, stream>>>(mag, wc, bc, wg, bg, out);
}
